// LightweightMambaDecoder_46978352284135
// MI455X (gfx1250) — matrix operand order repaired and checked by reading, not compiled, not run on hardware
//
#include <hip/hip_runtime.h>
#include <hip/hip_bf16.h>

typedef __attribute__((ext_vector_type(16))) _Float16 v16h;
typedef __attribute__((ext_vector_type(8)))  float    v8f;

// ---------------------------------------------------------------------------
// WMMA GEMM:  Y[batch] = W[batch%wmod] (MxK, row-major) * X[batch] (KxN) + epilogue
// One wave per 16x64 output tile: A fragment loaded once per K-block (4x
// float4 = global_load_b128 clause), then 4 v_wmma_f32_16x16x32_f16 against
// 4 B column-tiles (A reuse). Tiles beyond N are gated by a wave-uniform
// branch. Guarded path (K%32!=0 or partial M tile) uses clamped indices +
// select, loads gathered before conversion so they pipeline.
// Epilogue: optional bias (per-wb stride), optional BN (g,b,m,v), act:
//   0 = none, 1 = relu, 2 = softplus
// ---------------------------------------------------------------------------
__global__ void gemm_wmma_kernel(
    const float* __restrict__ Wm, const float* __restrict__ X, float* __restrict__ Y,
    int M, int N, int K,
    long long sW, long long sX, long long sY, int wmod,
    const float* __restrict__ bias, int biasStride,
    const float* __restrict__ bng, const float* __restrict__ bnb,
    const float* __restrict__ bnm, const float* __restrict__ bnv,
    int act)
{
    int batch = blockIdx.z;
    int wb = batch % wmod;
    Wm += (long long)wb * sW;
    X  += (long long)batch * sX;
    Y  += (long long)batch * sY;
    const float* bp = bias ? (bias + (long long)wb * biasStride) : nullptr;

    int n0   = blockIdx.x * 64;
    int m0   = blockIdx.y * 16;
    int lane = threadIdx.x & 31;
    int half = lane >> 4;     // which K-half this lane holds
    int lm   = lane & 15;
    int arow = m0 + lm;
    int nt   = (N - n0) >= 64 ? 4 : (N - n0) / 16;   // active 16-col tiles (uniform)

    const float* Xc = X + n0 + lm;   // this lane's B column (tile 0)
    v8f acc0 = {}, acc1 = {}, acc2 = {}, acc3 = {};

    bool fast = ((K & 31) == 0) && (m0 + 16 <= M);
    if (fast) {
        // A row pointer, pre-offset by this lane's K-half (16 floats = 64B aligned)
        const float* Wr = Wm + (long long)arow * K + 16 * half;
        for (int kb = 0; kb < K; kb += 32) {
            v16h a;
            float4 a0 = *(const float4*)(Wr + kb);        // K = kb+16h+0..3
            float4 a1 = *(const float4*)(Wr + kb + 4);    // K = kb+16h+4..7
            float4 a2 = *(const float4*)(Wr + kb + 8);    // K = kb+16h+8..11
            float4 a3 = *(const float4*)(Wr + kb + 12);   // K = kb+16h+12..15
            a[0]  = (_Float16)a0.x; a[1]  = (_Float16)a0.y;
            a[2]  = (_Float16)a0.z; a[3]  = (_Float16)a0.w;
            a[4]  = (_Float16)a1.x; a[5]  = (_Float16)a1.y;
            a[6]  = (_Float16)a1.z; a[7]  = (_Float16)a1.w;
            a[8]  = (_Float16)a2.x; a[9]  = (_Float16)a2.y;
            a[10] = (_Float16)a2.z; a[11] = (_Float16)a2.w;
            a[12] = (_Float16)a3.x; a[13] = (_Float16)a3.y;
            a[14] = (_Float16)a3.z; a[15] = (_Float16)a3.w;

            const float* Xk = Xc + (long long)(kb + 16 * half) * N;
            {
                float bt[16];
#pragma unroll
                for (int i = 0; i < 16; ++i) bt[i] = Xk[(long long)i * N];
                v16h bf;
#pragma unroll
                for (int i = 0; i < 16; ++i) bf[i] = (_Float16)bt[i];
                acc0 = __builtin_amdgcn_wmma_f32_16x16x32_f16(false, a, false, bf,
                                                              (short)0, acc0, false, false);
            }
            if (nt > 1) {
                float bt[16];
#pragma unroll
                for (int i = 0; i < 16; ++i) bt[i] = Xk[16 + (long long)i * N];
                v16h bf;
#pragma unroll
                for (int i = 0; i < 16; ++i) bf[i] = (_Float16)bt[i];
                acc1 = __builtin_amdgcn_wmma_f32_16x16x32_f16(false, a, false, bf,
                                                              (short)0, acc1, false, false);
            }
            if (nt > 2) {
                float bt[16];
#pragma unroll
                for (int i = 0; i < 16; ++i) bt[i] = Xk[32 + (long long)i * N];
                v16h bf;
#pragma unroll
                for (int i = 0; i < 16; ++i) bf[i] = (_Float16)bt[i];
                acc2 = __builtin_amdgcn_wmma_f32_16x16x32_f16(false, a, false, bf,
                                                              (short)0, acc2, false, false);
            }
            if (nt > 3) {
                float bt[16];
#pragma unroll
                for (int i = 0; i < 16; ++i) bt[i] = Xk[48 + (long long)i * N];
                v16h bf;
#pragma unroll
                for (int i = 0; i < 16; ++i) bf[i] = (_Float16)bt[i];
                acc3 = __builtin_amdgcn_wmma_f32_16x16x32_f16(false, a, false, bf,
                                                              (short)0, acc3, false, false);
            }
        }
    } else {
        // Guarded path: clamp indices, zero via select (no divergent branches)
        int arc = min(arow, M - 1);
        const float* Wr = Wm + (long long)arc * K;
        float amask = (arow < M) ? 1.0f : 0.0f;
        for (int kb = 0; kb < K; kb += 32) {
            v16h a;
            {
                float at[16];
#pragma unroll
                for (int i = 0; i < 16; ++i) {
                    int ka = kb + 16 * half + i;
                    at[i] = Wr[min(ka, K - 1)];
                }
#pragma unroll
                for (int i = 0; i < 16; ++i) {
                    int ka = kb + 16 * half + i;
                    a[i] = (_Float16)((ka < K) ? at[i] * amask : 0.0f);
                }
            }
#pragma unroll
            for (int j = 0; j < 4; ++j) {
                if (j < nt) {
                    const float* Xj = Xc + j * 16;
                    float bt[16];
#pragma unroll
                    for (int i = 0; i < 16; ++i) {
                        int kbi = kb + 16 * half + i;
                        bt[i] = Xj[(long long)min(kbi, K - 1) * N];
                    }
                    v16h bf;
#pragma unroll
                    for (int i = 0; i < 16; ++i) {
                        int kbi = kb + 16 * half + i;
                        bf[i] = (_Float16)((kbi < K) ? bt[i] : 0.0f);
                    }
                    v8f* accp = (j == 0) ? &acc0 : (j == 1) ? &acc1 : (j == 2) ? &acc2 : &acc3;
                    *accp = __builtin_amdgcn_wmma_f32_16x16x32_f16(false, a, false, bf,
                                                                   (short)0, *accp, false, false);
                }
            }
        }
    }

    // ---- epilogue + store (per active tile) ----
#pragma unroll
    for (int j = 0; j < 4; ++j) {
        if (j < nt) {
            v8f accv = (j == 0) ? acc0 : (j == 1) ? acc1 : (j == 2) ? acc2 : acc3;
            int col = n0 + j * 16 + lm;
#pragma unroll
            for (int v = 0; v < 8; ++v) {
                int row = m0 + v + 8 * half;   // C/D layout: lanes16-31 hold rows M+8
                if (row < M) {
                    float val = accv[v];
                    if (bp)  val += bp[row];
                    if (bng) {
                        float inv = bng[row] * rsqrtf(bnv[row] + 1e-5f);
                        val = (val - bnm[row]) * inv + bnb[row];
                    }
                    if (act == 1)      val = fmaxf(val, 0.0f);
                    else if (act == 2) val = (val > 20.0f) ? val : log1pf(expf(val));
                    Y[(long long)row * N + col] = val;
                }
            }
        }
    }
}

// ---------------------------------------------------------------------------
// Elementwise / support kernels
// ---------------------------------------------------------------------------
__global__ void ln2d_kernel(const float* __restrict__ x, float* __restrict__ y,
                            const float* __restrict__ g, const float* __restrict__ b,
                            int B, int C, int L)
{
    int idx = blockIdx.x * blockDim.x + threadIdx.x;
    if (idx >= B * L) return;
    int bb = idx / L, l = idx % L;
    const float* xp = x + ((long long)bb * C) * L + l;
    float mu = 0.f;
    for (int c = 0; c < C; ++c) mu += xp[(long long)c * L];
    mu /= C;
    float var = 0.f;
    for (int c = 0; c < C; ++c) { float d = xp[(long long)c * L] - mu; var += d * d; }
    var /= C;
    float rs = rsqrtf(var + 1e-5f);
    float* yp = y + ((long long)bb * C) * L + l;
    for (int c = 0; c < C; ++c)
        yp[(long long)c * L] = (xp[(long long)c * L] - mu) * rs * g[c] + b[c];
}

__global__ void dwconv3_kernel(const float* __restrict__ x, long long xBStride,
                               const float* __restrict__ w, const float* __restrict__ bias,
                               float* __restrict__ y, int B, int C, int H, int W)
{
    int idx = blockIdx.x * blockDim.x + threadIdx.x;
    int total = B * C * H * W;
    if (idx >= total) return;
    int wc = idx % W, h = (idx / W) % H, c = (idx / (W * H)) % C, b = idx / (W * H * C);
    const float* xp = x + (long long)b * xBStride + ((long long)c * H) * W;
    const float* wp = w + (long long)c * 9;
    float acc = bias[c];
    for (int kh = 0; kh < 3; ++kh) {
        int ih = h + kh - 1;
        if (ih < 0 || ih >= H) continue;
        for (int kw = 0; kw < 3; ++kw) {
            int iw = wc + kw - 1;
            if (iw < 0 || iw >= W) continue;
            acc += xp[ih * W + iw] * wp[kh * 3 + kw];
        }
    }
    y[idx] = acc / (1.0f + expf(-acc));   // silu
}

__global__ void build_xs_kernel(const float* __restrict__ xq, float* __restrict__ xs,
                                int B, int Di, int H, int W)
{
    int L = H * W;
    int idx = blockIdx.x * blockDim.x + threadIdx.x;
    int total = B * Di * L;
    if (idx >= total) return;
    int l = idx % L, d = (idx / L) % Di, b = idx / (L * Di);
    int h = l / W, w = l % W;
    float v = xq[idx];
    int lwh = w * H + h;
    long long b0 = ((long long)(b * 4 + 0) * Di + d) * L;
    long long b1 = ((long long)(b * 4 + 1) * Di + d) * L;
    long long b2 = ((long long)(b * 4 + 2) * Di + d) * L;
    long long b3 = ((long long)(b * 4 + 3) * Di + d) * L;
    xs[b0 + l]           = v;
    xs[b1 + lwh]         = v;
    xs[b2 + (L - 1 - l)] = v;
    xs[b3 + (L - 1 - lwh)] = v;
}

// Selective scan: one block per (b,k), one thread per channel d. 16-state h in
// registers; B/C vectors for the timestep staged through LDS.
__global__ void scan_kernel(const float* __restrict__ xs, const float* __restrict__ delta,
                            const float* __restrict__ xdbl, const float* __restrict__ A_log,
                            const float* __restrict__ Dp, float* __restrict__ ys,
                            int Di, int L)
{
    int blk = blockIdx.x;          // b*4 + k
    int k = blk & 3;
    int d = threadIdx.x;           // 0..Di-1
    long long base = ((long long)blk * Di + d) * L;
    long long b38  = (long long)blk * 38 * L;
    float A[16];
#pragma unroll
    for (int n = 0; n < 16; ++n)
        A[n] = -expf(A_log[((long long)k * Di + d) * 16 + n]);
    float Dv = Dp[k * Di + d];
    float h[16];
#pragma unroll
    for (int n = 0; n < 16; ++n) h[n] = 0.f;

    __shared__ float Bs[16], Cs[16];
    for (int t = 0; t < L; ++t) {
        if (d < 16)      Bs[d]      = xdbl[b38 + (long long)(6 + d) * L + t];
        else if (d < 32) Cs[d - 16] = xdbl[b38 + (long long)(22 + (d - 16)) * L + t];
        __syncthreads();
        float u  = xs[base + t];
        float dt = delta[base + t];
        float du = dt * u;
        float accv = 0.f;
#pragma unroll
        for (int n = 0; n < 16; ++n) {
            h[n] = h[n] * expf(dt * A[n]) + du * Bs[n];
            accv += h[n] * Cs[n];
        }
        ys[base + t] = accv + Dv * u;   // ys may alias delta (read-before-write per t)
        __syncthreads();
    }
}

__global__ void combine_ys_kernel(const float* __restrict__ ys, float* __restrict__ ysum,
                                  int B, int Di, int H, int W)
{
    int L = H * W;
    int idx = blockIdx.x * blockDim.x + threadIdx.x;
    int total = B * Di * L;
    if (idx >= total) return;
    int l = idx % L, d = (idx / L) % Di, b = idx / (L * Di);
    int h = l / W, w = l % W;
    int lwh = w * H + h;
    long long b0 = ((long long)(b * 4 + 0) * Di + d) * L;
    long long b1 = ((long long)(b * 4 + 1) * Di + d) * L;
    long long b2 = ((long long)(b * 4 + 2) * Di + d) * L;
    long long b3 = ((long long)(b * 4 + 3) * Di + d) * L;
    ysum[idx] = ys[b0 + l] + ys[b2 + (L - 1 - l)]
              + ys[b1 + lwh] + ys[b3 + (L - 1 - lwh)];
}

__global__ void ln_silu_kernel(const float* __restrict__ y, const float* __restrict__ zbase,
                               float* __restrict__ out, const float* __restrict__ g,
                               const float* __restrict__ b, int B, int Di, int L,
                               long long zBStride)
{
    int idx = blockIdx.x * blockDim.x + threadIdx.x;
    if (idx >= B * L) return;
    int bb = idx / L, l = idx % L;
    const float* yp = y + ((long long)bb * Di) * L + l;
    float mu = 0.f;
    for (int d = 0; d < Di; ++d) mu += yp[(long long)d * L];
    mu /= Di;
    float var = 0.f;
    for (int d = 0; d < Di; ++d) { float t = yp[(long long)d * L] - mu; var += t * t; }
    var /= Di;
    float rs = rsqrtf(var + 1e-5f);
    const float* zp = zbase + (long long)bb * zBStride + l;
    float* op = out + ((long long)bb * Di) * L + l;
    for (int d = 0; d < Di; ++d) {
        float v  = (yp[(long long)d * L] - mu) * rs * g[d] + b[d];
        float zv = zp[(long long)d * L];
        op[(long long)d * L] = v * (zv / (1.0f + expf(-zv)));
    }
}

__global__ void residual_kernel(float* __restrict__ x, const float* __restrict__ xn,
                                const float* __restrict__ ss, int total)
{
    int idx = blockIdx.x * blockDim.x + threadIdx.x;
    if (idx >= total) return;
    x[idx] = x[idx] + 0.1f * (xn[idx] + ss[idx]);
}

__global__ void resize_bilinear_kernel(const float* __restrict__ x, float* __restrict__ y,
                                       int B, int C, int H, int W, int OH, int OW)
{
    int idx = blockIdx.x * blockDim.x + threadIdx.x;
    int total = B * C * OH * OW;
    if (idx >= total) return;
    int ox = idx % OW, oy = (idx / OW) % OH;
    int c = (idx / (OW * OH)) % C, b = idx / (OW * OH * C);
    float fy = (OH > 1) ? oy * (float)(H - 1) / (float)(OH - 1) : 0.f;
    float fx = (OW > 1) ? ox * (float)(W - 1) / (float)(OW - 1) : 0.f;
    int y0 = (int)floorf(fy); float wy = fy - y0; int y1 = min(y0 + 1, H - 1);
    int x0 = (int)floorf(fx); float wx = fx - x0; int x1 = min(x0 + 1, W - 1);
    const float* xp = x + (((long long)b * C + c) * H) * W;
    float top = xp[y0 * W + x0] * (1 - wx) + xp[y0 * W + x1] * wx;
    float bot = xp[y1 * W + x0] * (1 - wx) + xp[y1 * W + x1] * wx;
    y[idx] = top * (1 - wy) + bot * wy;
}

__global__ void mean_hw_kernel(const float* __restrict__ x, float* __restrict__ y,
                               int B, int C, int L)
{
    int idx = blockIdx.x * blockDim.x + threadIdx.x;
    if (idx >= B * C) return;
    const float* xp = x + (long long)idx * L;
    float s = 0.f;
    for (int l = 0; l < L; ++l) s += xp[l];
    y[idx] = s / L;
}

__global__ void attn_kernel(const float* __restrict__ pooled, const float* __restrict__ w1,
                            const float* __restrict__ b1, const float* __restrict__ w2,
                            const float* __restrict__ b2, const float* __restrict__ basew,
                            float* __restrict__ wout, int B, int C, int Hd)
{
    int b = threadIdx.x;
    if (b >= B) return;
    float hid[32];
    for (int j = 0; j < Hd; ++j) {
        float s = b1[j];
        for (int c = 0; c < C; ++c) s += w1[j * C + c] * pooled[b * C + c];
        hid[j] = fmaxf(s, 0.f);
    }
    float lg[3]; float mx = -1e30f;
    for (int i = 0; i < 3; ++i) {
        float s = b2[i];
        for (int j = 0; j < Hd; ++j) s += w2[i * Hd + j] * hid[j];
        lg[i] = s; mx = fmaxf(mx, s);
    }
    float den = 0.f;
    for (int i = 0; i < 3; ++i) { lg[i] = expf(lg[i] - mx); den += lg[i]; }
    for (int i = 0; i < 3; ++i) wout[b * 3 + i] = 0.7f * basew[i] + 0.3f * lg[i] / den;
}

__global__ void weighted_kernel(const float* __restrict__ f, const float* __restrict__ wattn,
                                float* __restrict__ wcat, int B, int C, int L, int which)
{
    int idx = blockIdx.x * blockDim.x + threadIdx.x;
    int total = B * C * L;
    if (idx >= total) return;
    int l = idx % L, c = (idx / L) % C, b = idx / (L * C);
    long long o = (((long long)b * 3 * C) + which * C + c) * L + l;
    wcat[o] = f[idx] * wattn[b * 3 + which];
}

__global__ void conv3x3_bn_kernel(const float* __restrict__ x, const float* __restrict__ w,
                                  const float* __restrict__ bias,
                                  const float* __restrict__ bg, const float* __restrict__ bb,
                                  const float* __restrict__ bm, const float* __restrict__ bv,
                                  int relu, float* __restrict__ y,
                                  int B, int Cin, int Cout, int H, int W,
                                  int groups, int dil, int pad)
{
    int idx = blockIdx.x * blockDim.x + threadIdx.x;
    int total = B * Cout * H * W;
    if (idx >= total) return;
    int wc = idx % W, h = (idx / W) % H, co = (idx / (W * H)) % Cout, b = idx / (W * H * Cout);
    int cpgIn = Cin / groups, cpgOut = Cout / groups;
    int gI = co / cpgOut;
    float acc = bias[co];
    for (int ci = 0; ci < cpgIn; ++ci) {
        int c = gI * cpgIn + ci;
        const float* xp = x + (((long long)b * Cin + c) * H) * W;
        const float* wp = w + ((long long)co * cpgIn + ci) * 9;
        for (int kh = 0; kh < 3; ++kh) {
            int ih = h - pad + kh * dil;
            if (ih < 0 || ih >= H) continue;
            for (int kw = 0; kw < 3; ++kw) {
                int iw = wc - pad + kw * dil;
                if (iw < 0 || iw >= W) continue;
                acc += xp[ih * W + iw] * wp[kh * 3 + kw];
            }
        }
    }
    float inv = bg[co] * rsqrtf(bv[co] + 1e-5f);
    acc = (acc - bm[co]) * inv + bb[co];
    if (relu) acc = fmaxf(acc, 0.f);
    y[idx] = acc;
}

__global__ void gate_kernel(const float* __restrict__ gpool, const float* __restrict__ g1w,
                            const float* __restrict__ g1b, const float* __restrict__ g2w,
                            const float* __restrict__ g2b, float* __restrict__ gw, int B)
{
    int b = threadIdx.x;
    if (b >= B) return;
    float hid[24];
    for (int j = 0; j < 24; ++j) {
        float s = g1b[j];
        for (int c = 0; c < 96; ++c) s += g1w[j * 96 + c] * gpool[b * 96 + c];
        hid[j] = fmaxf(s, 0.f);
    }
    for (int c = 0; c < 96; ++c) {
        float s = g2b[c];
        for (int j = 0; j < 24; ++j) s += g2w[c * 24 + j] * hid[j];
        gw[b * 96 + c] = 1.0f / (1.0f + expf(-s));
    }
}

__global__ void ctx_add_kernel(const float* __restrict__ fused, const float* __restrict__ e3,
                               const float* __restrict__ gw, float* __restrict__ wcat,
                               int B, int C, int L)
{
    int idx = blockIdx.x * blockDim.x + threadIdx.x;
    int total = B * C * L;
    if (idx >= total) return;
    int l = idx % L, c = (idx / L) % C, b = idx / (L * C);
    float ctx = fused[idx] + e3[idx] * gw[b * C + c];
    long long base = ((long long)b * 3 * C) * L;
    wcat[base + ((long long)(0 * C + c)) * L + l] += ctx;
    wcat[base + ((long long)(1 * C + c)) * L + l] += ctx;
    wcat[base + ((long long)(2 * C + c)) * L + l] += ctx;
}

__global__ void final_proj_kernel(const float* __restrict__ t2, const float* __restrict__ fw,
                                  const float* __restrict__ fb, float* __restrict__ out,
                                  int B, int C, int L)
{
    int idx = blockIdx.x * blockDim.x + threadIdx.x;
    if (idx >= B * L) return;
    int b = idx / L, l = idx % L;
    float acc = fb[0];
    for (int c = 0; c < C; ++c) acc += fw[c] * t2[((long long)b * C + c) * L + l];
    out[idx] = acc;
}

// ---------------------------------------------------------------------------
// Host side
// ---------------------------------------------------------------------------
namespace {

constexpr int NB = 8;     // batch
constexpr int DIMC = 96;  // DIM
constexpr int DI = 192;   // DIM * SSM_RATIO

inline void launch_gemm(hipStream_t st, const float* Wm, const float* X, float* Y,
                        int M, int N, int K, int batches,
                        long long sW, long long sX, long long sY, int wmod,
                        const float* bias = nullptr, int biasStride = 0,
                        const float* g = nullptr, const float* b = nullptr,
                        const float* m = nullptr, const float* v = nullptr,
                        int act = 0)
{
    dim3 grid((unsigned)((N + 63) / 64), (unsigned)((M + 15) / 16), (unsigned)batches);
    gemm_wmma_kernel<<<grid, dim3(32, 1, 1), 0, st>>>(
        Wm, X, Y, M, N, K, sW, sX, sY, wmod, bias, biasStride, g, b, m, v, act);
}

#define EW(kern, total, ...) do { int _t = (int)(total); \
    if (_t > 0) kern<<<(_t + 255) / 256, 256, 0, stream>>>(__VA_ARGS__); } while (0)

struct EnhWs {
    float *xn, *sin_, *xz, *xq, *xs, *xdbl, *delta, *ymul, *ssout;
};

inline void run_enhancer(hipStream_t stream, const float* const* P, int pb,
                         float* x, int H, int W, const EnhWs& ws)
{
    int L = H * W;
    // xn = ln(x, in_g, in_b); sin = ln(xn, ln1_g, ln1_b)
    EW(ln2d_kernel, NB * L, x, ws.xn, P[pb + 0], P[pb + 1], NB, DIMC, L);
    EW(ln2d_kernel, NB * L, ws.xn, ws.sin_, P[pb + 2], P[pb + 3], NB, DIMC, L);
    // xz = in_proj(sin): (2*Di x 96) GEMM per batch
    launch_gemm(stream, P[pb + 4], ws.sin_, ws.xz, 2 * DI, L, DIMC, NB,
                0, (long long)DIMC * L, (long long)2 * DI * L, 1);
    // xq = silu(dwconv3x3(xz[:, :Di]))
    EW(dwconv3_kernel, NB * DI * L, ws.xz, (long long)2 * DI * L,
       P[pb + 5], P[pb + 6], ws.xq, NB, DI, H, W);
    // xs: 4 scan orders
    EW(build_xs_kernel, NB * DI * L, ws.xq, ws.xs, NB, DI, H, W);
    // x_dbl = x_proj_w[k] (38 x 192) * xs[b,k]
    launch_gemm(stream, P[pb + 7], ws.xs, ws.xdbl, 38, L, DI, NB * 4,
                38LL * DI, (long long)DI * L, 38LL * L, 4);
    // delta = softplus(dt_w[k] (192 x 6) * dts + dt_b[k])
    launch_gemm(stream, P[pb + 8], ws.xdbl, ws.delta, DI, L, 6, NB * 4,
                (long long)DI * 6, 38LL * L, (long long)DI * L, 4,
                P[pb + 9], DI, nullptr, nullptr, nullptr, nullptr, 2);
    // selective scan (ys written in place over delta)
    scan_kernel<<<NB * 4, DI, 0, stream>>>(ws.xs, ws.delta, ws.xdbl,
                                           P[pb + 10], P[pb + 11], ws.delta, DI, L);
    // recombine 4 directions -> ysum (reuse xq buffer)
    EW(combine_ys_kernel, NB * DI * L, ws.delta, ws.xq, NB, DI, H, W);
    // out_norm layernorm * silu(z)
    EW(ln_silu_kernel, NB * L, ws.xq, ws.xz + (long long)DI * L, ws.ymul,
       P[pb + 12], P[pb + 13], NB, DI, L, (long long)2 * DI * L);
    // out_proj (96 x 192)
    launch_gemm(stream, P[pb + 14], ws.ymul, ws.ssout, DIMC, L, DI, NB,
                0, (long long)DI * L, (long long)DIMC * L, 1);
    // x = x + 0.1*(xn + ssout)
    EW(residual_kernel, NB * DIMC * L, x, ws.xn, ws.ssout, NB * DIMC * L);
}

} // namespace

extern "C" void kernel_launch(void* const* d_in, const int* in_sizes, int n_in,
                              void* d_out, int out_size, void* d_ws, size_t ws_size,
                              hipStream_t stream)
{
    (void)in_sizes; (void)n_in; (void)out_size;
    const float* P[128];
    for (int i = 0; i < n_in && i < 128; ++i) P[i] = (const float*)d_in[i];
    float* out = (float*)d_out;

    const int L1 = 48 * 48, L2 = 24 * 24, L3 = 12 * 12;

    // ---- workspace bump allocation (floats) ----
    float* wsf = (float*)d_ws;
    size_t off = 0;
    auto alloc = [&](size_t n) { float* p = wsf + off; off += n; return p; };

    float* f3p   = alloc((size_t)NB * DIMC * L3);
    float* f2p   = alloc((size_t)NB * DIMC * L2);
    float* f1p   = alloc((size_t)NB * DIMC * L1);
    float* fd2   = alloc((size_t)NB * DIMC * L1);
    float* fd3   = alloc((size_t)NB * DIMC * L1);
    float* wcat  = alloc((size_t)NB * 3 * DIMC * L1);
    float* fused = alloc((size_t)NB * DIMC * L1);
    float* e1    = alloc((size_t)NB * 48 * L1);
    float* e2    = alloc((size_t)NB * 48 * L1);
    float* e3    = alloc((size_t)NB * DIMC * L1);
    float* pooled= alloc(64 * 2);
    float* gpool = alloc(64 * 2);
    float* wattn = alloc(64);
    float* gw    = alloc(64 * 2);
    float* t1    = alloc((size_t)NB * DIMC * L1);
    float* t2    = alloc((size_t)NB * 48 * L1);

    EnhWs ews;
    ews.xn    = alloc((size_t)NB * DIMC * L1);
    ews.sin_  = alloc((size_t)NB * DIMC * L1);
    ews.xz    = alloc((size_t)NB * 2 * DI * L1);
    ews.xq    = alloc((size_t)NB * DI * L1);
    ews.xs    = alloc((size_t)NB * 4 * DI * L1);
    ews.xdbl  = alloc((size_t)NB * 4 * 38 * L1);
    ews.delta = alloc((size_t)NB * 4 * DI * L1);
    ews.ymul  = alloc((size_t)NB * DI * L1);
    ews.ssout = alloc((size_t)NB * DIMC * L1);

    if (off * sizeof(float) > ws_size) return;   // workspace too small: bail

    // ---- input projections: relu(bn(conv1x1 + bias)) ----
    launch_gemm(stream, P[3], P[0], f3p, DIMC, L3, 384, NB,
                0, 384LL * L3, (long long)DIMC * L3, 1,
                P[4], 0, P[5], P[6], P[7], P[8], 1);
    launch_gemm(stream, P[9], P[1], f2p, DIMC, L2, 192, NB,
                0, 192LL * L2, (long long)DIMC * L2, 1,
                P[10], 0, P[11], P[12], P[13], P[14], 1);
    launch_gemm(stream, P[15], P[2], f1p, DIMC, L1, 96, NB,
                0, 96LL * L1, (long long)DIMC * L1, 1,
                P[16], 0, P[17], P[18], P[19], P[20], 1);

    // ---- enhancers (SS2D Mamba blocks) ----
    run_enhancer(stream, P, 21, f3p, 12, 12, ews);
    run_enhancer(stream, P, 36, f2p, 24, 24, ews);
    run_enhancer(stream, P, 51, f1p, 48, 48, ews);

    // ---- upsample to 48x48 (align-corners bilinear) ----
    EW(resize_bilinear_kernel, NB * DIMC * L1, f3p, fd3, NB, DIMC, 12, 12, 48, 48);
    EW(resize_bilinear_kernel, NB * DIMC * L1, f2p, fd2, NB, DIMC, 24, 24, 48, 48);

    // ---- adaptive scale-attention weights ----
    EW(mean_hw_kernel, NB * DIMC, f1p, pooled, NB, DIMC, L1);
    attn_kernel<<<1, 32, 0, stream>>>(pooled, P[66], P[67], P[68], P[69], P[70],
                                      wattn, NB, 96, 24);
    EW(weighted_kernel, NB * DIMC * L1, f1p, wattn, wcat, NB, DIMC, L1, 0);
    EW(weighted_kernel, NB * DIMC * L1, fd2, wattn, wcat, NB, DIMC, L1, 1);
    EW(weighted_kernel, NB * DIMC * L1, fd3, wattn, wcat, NB, DIMC, L1, 2);

    // ---- fused = relu(bn(conv1x1(wcat))) ----
    launch_gemm(stream, P[71], wcat, fused, DIMC, L1, 3 * DIMC, NB,
                0, (long long)3 * DIMC * L1, (long long)DIMC * L1, 1,
                P[72], 0, P[73], P[74], P[75], P[76], 1);

    // ---- multi-scale modulation (mms) ----
    EW(conv3x3_bn_kernel, NB * 48 * L1, fused, P[77], P[78],
       P[79], P[80], P[81], P[82], 1, e1, NB, 96, 48, 48, 48, 24, 1, 1);
    EW(conv3x3_bn_kernel, NB * 48 * L1, e1, P[83], P[84],
       P[85], P[86], P[87], P[88], 1, e2, NB, 48, 48, 48, 48, 1, 2, 2);
    launch_gemm(stream, P[89], e2, e3, DIMC, L1, 48, NB,
                0, 48LL * L1, (long long)DIMC * L1, 1,
                P[90], 0, P[91], P[92], P[93], P[94], 0);
    EW(mean_hw_kernel, NB * DIMC, fused, gpool, NB, DIMC, L1);
    gate_kernel<<<1, 32, 0, stream>>>(gpool, P[95], P[96], P[97], P[98], gw, NB);
    // ctx = fused + e3*gw ; enhanced = weighted + ctx (in place on wcat)
    EW(ctx_add_kernel, NB * DIMC * L1, fused, e3, gw, wcat, NB, DIMC, L1);

    // ---- prediction head ----
    launch_gemm(stream, P[99], wcat, t1, DIMC, L1, 3 * DIMC, NB,
                0, (long long)3 * DIMC * L1, (long long)DIMC * L1, 1,
                P[100], 0, nullptr, nullptr, nullptr, nullptr, 1);
    launch_gemm(stream, P[101], t1, t2, 48, L1, DIMC, NB,
                0, (long long)DIMC * L1, 48LL * L1, 1,
                P[102], 0, nullptr, nullptr, nullptr, nullptr, 1);
    EW(final_proj_kernel, NB * L1, t2, P[103], P[104], out, NB, 48, L1);
}
